// RGLRU_20315195310625
// MI455X (gfx1250) — matrix operand order repaired and checked by reading, not compiled, not run on hardware
//
#include <hip/hip_runtime.h>
#include <hip/hip_bf16.h>

// Problem constants (from reference): B=4, L=4096, W=2560, H=10, BLK=256
#define BQ   4
#define LQ   4096
#define WQ   2560
#define HQ   10
#define BLKQ 256

// LDS row stride for transposed weights (pad 256 -> 264 bf16 to break bank conflicts)
#define WST  264

// Scan chunking
#define CHK  64              // chunks per sequence
#define CLEN (LQ / CHK)      // 64 steps per chunk

typedef __attribute__((ext_vector_type(16))) __bf16 v16bf;
typedef __attribute__((ext_vector_type(8)))  float  v8f;

__device__ __forceinline__ float sigmoidf_fast(float v) {
    return 1.0f / (1.0f + __expf(-v));
}

// ---------------------------------------------------------------------------
// Phase 1: block-diagonal GEMMs (bf16 WMMA) + gate math.
// grid = (256 row-tiles of 64, H heads), block = 256 threads (8 wave32).
// Writes normalized_x into h_out (d_out's h region) and a_t into at_out (ws).
// ---------------------------------------------------------------------------
__global__ __launch_bounds__(256) void rglru_gates_wmma(
    const float* __restrict__ x,        // (B*L, W)
    const int*   __restrict__ segpos,   // (B*L)
    const float* __restrict__ wgx,      // (H, 256, 256)
    const float* __restrict__ bgx,      // (H, 256)
    const float* __restrict__ wga,      // (H, 256, 256)
    const float* __restrict__ bga,      // (H, 256)
    const float* __restrict__ apar,     // (W)
    float* __restrict__ h_out,          // (B*L, W)  <- normalized_x
    float* __restrict__ at_out)         // (B*L, W)  <- a_t
{
    // 2 x 256x264 bf16 = 264 KB (fits 320 KB WGP LDS), stored TRANSPOSED:
    // sW[g][j*WST + i] = w[g][h][i][j]  so a B-fragment (fixed column j,
    // 16 consecutive K=i) is one contiguous 32-byte LDS read.
    __shared__ __bf16 sW[2][256 * WST];

    const int h     = blockIdx.y;
    const int mtile = blockIdx.x;       // 64-row tile of the 16384 rows
    const int tid   = threadIdx.x;

    const float* g0 = wgx + (size_t)h * (BLKQ * BLKQ);
    const float* g1 = wga + (size_t)h * (BLKQ * BLKQ);
    for (int idx = tid; idx < BLKQ * BLKQ; idx += 256) {
        const int i = idx >> 8;         // input dim (K)
        const int j = idx & 255;        // output dim (N)
        sW[0][j * WST + i] = (__bf16)g0[idx];
        sW[1][j * WST + i] = (__bf16)g1[idx];
    }
    __syncthreads();

    const int wave = tid >> 5;          // wave32
    const int lane = tid & 31;
    const int rg   = wave & 3;          // row group 0..3 (16 rows each)
    const int chh  = wave >> 2;         // column half 0..1 (8 col tiles each)
    const int hl   = lane >> 4;         // lane half 0/1
    const int l16  = lane & 15;

    const int mbase = mtile * 64 + rg * 16;

    v8f accx[8], acca[8];
    #pragma unroll
    for (int t = 0; t < 8; ++t) { accx[t] = {}; acca[t] = {}; }

    // A-fragment source row for this lane (16-bit A layout: lane%16 = M)
    const int    arow = mbase + l16;
    const float* xrow = x + (size_t)arow * WQ + h * BLKQ;

    #pragma unroll 1
    for (int kk = 0; kk < 8; ++kk) {            // K = 256 in chunks of 32
        // Build 16x32 bf16 A fragment in contiguous k order (same as B):
        // lane<16: K = kbase+{0..15}; lane>=16: +16 offset.
        const int kbase = kk * 32 + hl * 16;
        v16bf afrag;
        #pragma unroll
        for (int i = 0; i < 8; ++i) {
            afrag[i]     = (__bf16)xrow[kbase + i];
            afrag[i + 8] = (__bf16)xrow[kbase + 8 + i];
        }
        // B fragment: lane -> column j = tile*16 + lane%16, K run of 16
        const int koff = kk * 32 + hl * 16;
        #pragma unroll
        for (int t = 0; t < 8; ++t) {
            const int j = (chh * 8 + t) * 16 + l16;
            v16bf b0 = *(const v16bf*)(&sW[0][j * WST + koff]);
            v16bf b1 = *(const v16bf*)(&sW[1][j * WST + koff]);
            accx[t] = __builtin_amdgcn_wmma_f32_16x16x32_bf16(
                false, afrag, false, b0, (short)0, accx[t], false, false);
            acca[t] = __builtin_amdgcn_wmma_f32_16x16x32_bf16(
                false, afrag, false, b1, (short)0, acca[t], false, false);
        }
    }

    // Epilogue. D layout: lane holds column N = lane%16, rows M = v + 8*(lane/16).
    #pragma unroll
    for (int t = 0; t < 8; ++t) {
        const int colh  = (chh * 8 + t) * 16 + l16;   // column within head
        const int wfull = h * BLKQ + colh;            // global channel
        const float bx = bgx[h * BLKQ + colh];
        const float ba = bga[h * BLKQ + colh];
        const float sp = log1pf(__expf(apar[wfull])); // softplus(a_param)
        #pragma unroll
        for (int v = 0; v < 8; ++v) {
            const int r = mbase + hl * 8 + v;         // global row in [0, B*L)
            const bool reset = (segpos[r] == 0);
            const float gx    = sigmoidf_fast(accx[t][v] + bx);
            const float ga    = sigmoidf_fast(acca[t][v] + ba);
            const float log_a = -8.0f * ga * sp;
            const float a     = __expf(log_a);
            const float a2    = __expf(2.0f * log_a);
            const float xv    = x[(size_t)r * WQ + wfull];
            const float mult  = reset ? 1.0f : sqrtf(fmaxf(1.0f - a2, 0.0f));
            h_out [(size_t)r * WQ + wfull] = xv * gx * mult;   // normalized_x
            at_out[(size_t)r * WQ + wfull] = reset ? 0.0f : a; // a_t
        }
    }
}

// ---------------------------------------------------------------------------
// Phase 2a: chunk-local sequential scan, in place.
// After this: h[idx] = chunk-local scan value S_t, at[idx] = chunk-local
// cumulative product A_t. Chunk summaries written to sumA/sumS.
// ---------------------------------------------------------------------------
__global__ __launch_bounds__(256) void rglru_scan_chunk(
    float* __restrict__ h, float* __restrict__ at,
    float* __restrict__ sumA, float* __restrict__ sumS)
{
    const int w = blockIdx.x * 256 + threadIdx.x;
    const int c = blockIdx.y;
    const int b = blockIdx.z;
    float A = 1.0f, S = 0.0f;
    const size_t base = ((size_t)b * LQ + (size_t)c * CLEN) * WQ + w;
    for (int t = 0; t < CLEN; ++t) {
        const size_t idx = base + (size_t)t * WQ;
        if (t + 8 < CLEN) {       // hide latency of the strided stream
            __builtin_prefetch(at + idx + (size_t)8 * WQ, 0, 1);
            __builtin_prefetch(h  + idx + (size_t)8 * WQ, 0, 1);
        }
        const float a  = at[idx];
        const float nx = h[idx];
        S = fmaf(a, S, nx);
        A *= a;
        h[idx]  = S;
        at[idx] = A;
    }
    const size_t si = ((size_t)b * CHK + c) * WQ + w;
    sumA[si] = A;
    sumS[si] = S;
}

// ---------------------------------------------------------------------------
// Phase 2b: scan the CHK chunk summaries per channel; carry[c] = global h at
// the entry of chunk c (carry[0] = h0, which also reproduces the a_cum*h0 term).
// ---------------------------------------------------------------------------
__global__ __launch_bounds__(256) void rglru_scan_carry(
    const float* __restrict__ sumA, const float* __restrict__ sumS,
    const float* __restrict__ h0, float* __restrict__ carry)
{
    const int w = blockIdx.x * 256 + threadIdx.x;
    const int b = blockIdx.z;
    float cr = h0[(size_t)b * WQ + w];
    for (int c = 0; c < CHK; ++c) {
        const size_t si = ((size_t)b * CHK + c) * WQ + w;
        carry[si] = cr;
        cr = fmaf(sumA[si], cr, sumS[si]);
    }
}

// ---------------------------------------------------------------------------
// Phase 2c: apply carries: h_t = S_t + A_t * carry[chunk(t)]; emit last_h.
// ---------------------------------------------------------------------------
__global__ __launch_bounds__(256) void rglru_scan_apply(
    float* __restrict__ h, const float* __restrict__ at,
    const float* __restrict__ carry, float* __restrict__ last)
{
    const int w = blockIdx.x * 256 + threadIdx.x;
    const int t = blockIdx.y;
    const int b = blockIdx.z;
    const int c = t / CLEN;
    const size_t idx = ((size_t)b * LQ + t) * WQ + w;
    const float v = fmaf(at[idx], carry[((size_t)b * CHK + c) * WQ + w], h[idx]);
    h[idx] = v;
    if (t == LQ - 1) last[(size_t)b * WQ + w] = v;
}

// ---------------------------------------------------------------------------
extern "C" void kernel_launch(void* const* d_in, const int* in_sizes, int n_in,
                              void* d_out, int out_size, void* d_ws, size_t ws_size,
                              hipStream_t stream)
{
    const float* x      = (const float*)d_in[0];
    const int*   segpos = (const int*)  d_in[1];
    const float* wgx    = (const float*)d_in[2];
    const float* bgx    = (const float*)d_in[3];
    const float* wga    = (const float*)d_in[4];
    const float* bga    = (const float*)d_in[5];
    const float* apar   = (const float*)d_in[6];
    const float* h0     = (const float*)d_in[7];

    float* h_out  = (float*)d_out;                       // (B, L, W)
    float* last_h = h_out + (size_t)BQ * LQ * WQ;        // (B, W)

    // Workspace carve-up
    float* at_ws = (float*)d_ws;                                  // B*L*W
    float* sumA  = at_ws + (size_t)BQ * LQ * WQ;                  // B*CHK*W
    float* sumS  = sumA  + (size_t)BQ * CHK * WQ;                 // B*CHK*W
    float* carry = sumS  + (size_t)BQ * CHK * WQ;                 // B*CHK*W

    // Phase 1: gates via bf16 WMMA (16384 rows / 64 per tile = 256 tiles, 10 heads)
    rglru_gates_wmma<<<dim3((BQ * LQ) / 64, HQ), 256, 0, stream>>>(
        x, segpos, wgx, bgx, wga, bga, apar, h_out, at_ws);

    // Phase 2: chunked associative scan over L
    rglru_scan_chunk<<<dim3(WQ / 256, CHK, BQ), 256, 0, stream>>>(
        h_out, at_ws, sumA, sumS);
    rglru_scan_carry<<<dim3(WQ / 256, 1, BQ), 256, 0, stream>>>(
        sumA, sumS, h0, carry);
    rglru_scan_apply<<<dim3(WQ / 256, LQ, BQ), 256, 0, stream>>>(
        h_out, at_ws, carry, last_h);
}
